// Gemma3DecoderLayer_52673478918589
// MI455X (gfx1250) — compile-verified, matrix operand order checked by reading, not run on hardware
//
#include <hip/hip_runtime.h>
#include <hip/hip_bf16.h>

typedef __attribute__((ext_vector_type(16))) __bf16 v16bf;
typedef __attribute__((ext_vector_type(8)))  float  v8f;

#define BB   2
#define SS   2048
#define HH   2560
#define NH_  8
#define NKV_ 4
#define HD_  256
#define INTER_ 10240
#define SW_  1024
#define SCALE_ 0.0625f

union Frag { v16bf v; uint4 q[2]; };

__device__ __forceinline__ unsigned short f2bf(float f) {
  unsigned int u = __float_as_uint(f);
  unsigned int r = u + 0x7FFFu + ((u >> 16) & 1u);
  return (unsigned short)(r >> 16);
}

__device__ __forceinline__ v8f vzero8() {
  v8f z = {0.f,0.f,0.f,0.f,0.f,0.f,0.f,0.f};
  return z;
}

__device__ __forceinline__ v8f wmma_bf16(v16bf a, v16bf b, v8f c) {
  return __builtin_amdgcn_wmma_f32_16x16x32_bf16(false, a, false, b, (short)0, c, false, false);
}

__device__ __forceinline__ float gelu_tanh(float g) {
  return 0.5f * g * (1.0f + tanhf(0.7978845608f * (g + 0.044715f * g * g * g)));
}

// async global->LDS copy of 32 contiguous bytes per lane (CDNA5 ASYNCcnt path)
__device__ __forceinline__ void async_copy32(unsigned lds_addr, const void* gaddr) {
  asm volatile(
      "global_load_async_to_lds_b128 %0, %1, off\n\t"
      "global_load_async_to_lds_b128 %0, %1, off offset:16"
      :: "v"(lds_addr), "v"((unsigned long long)gaddr) : "memory");
}
// async global->LDS copy of 64 contiguous bytes per lane
__device__ __forceinline__ void async_copy64(unsigned lds_addr, const void* gaddr) {
  asm volatile(
      "global_load_async_to_lds_b128 %0, %1, off\n\t"
      "global_load_async_to_lds_b128 %0, %1, off offset:16\n\t"
      "global_load_async_to_lds_b128 %0, %1, off offset:32\n\t"
      "global_load_async_to_lds_b128 %0, %1, off offset:48"
      :: "v"(lds_addr), "v"((unsigned long long)gaddr) : "memory");
}
__device__ __forceinline__ void wait_async() {
  asm volatile("s_wait_asynccnt 0x0" ::: "memory");
}
__device__ __forceinline__ unsigned lds_off(const void* p) {
  // generic LDS pointers carry the LDS byte offset in the low 32 bits
  return (unsigned)(unsigned long long)p;
}

__device__ __forceinline__ float block_sum256(float v) {
  __shared__ float red[8];
  #pragma unroll
  for (int off = 16; off > 0; off >>= 1) v += __shfl_xor(v, off);
  if ((threadIdx.x & 31) == 0) red[threadIdx.x >> 5] = v;
  __syncthreads();
  if (threadIdx.x == 0) {
    float s = 0.f;
    #pragma unroll
    for (int i = 0; i < 8; ++i) s += red[i];
    red[0] = s;
  }
  __syncthreads();
  return red[0];
}

// ---------------- RMSNorm -> bf16 ----------------
__global__ __launch_bounds__(256) void rms_to_bf16(const float* __restrict__ in,
                                                   const float* __restrict__ w,
                                                   unsigned short* __restrict__ out, int n) {
  size_t row = blockIdx.x;
  const float* x = in + row * n;
  float ss = 0.f;
  for (int i = threadIdx.x; i < n; i += 256) { float v = x[i]; ss += v * v; }
  float tot = block_sum256(ss);
  float inv = rsqrtf(tot / (float)n + 1e-6f);
  unsigned short* o = out + row * n;
  for (int i = threadIdx.x; i < n; i += 256)
    o[i] = f2bf((1.f + w[i]) * x[i] * inv);
}

// ---------------- out = res + RMSNorm(in) (fp32) ----------------
__global__ __launch_bounds__(256) void rms_add(const float* __restrict__ in,
                                               const float* __restrict__ w,
                                               const float* __restrict__ res,
                                               float* __restrict__ out, int n) {
  size_t row = blockIdx.x;
  const float* x = in + row * n;
  float ss = 0.f;
  for (int i = threadIdx.x; i < n; i += 256) { float v = x[i]; ss += v * v; }
  float tot = block_sum256(ss);
  float inv = rsqrtf(tot / (float)n + 1e-6f);
  const float* r = res + row * n;
  float* o = out + row * n;
  for (int i = threadIdx.x; i < n; i += 256)
    o[i] = r[i] + (1.f + w[i]) * x[i] * inv;
}

// ---------------- per-head RMSNorm + RoPE -> bf16 (one 256-wide head row per block) ----------------
__global__ __launch_bounds__(256) void qk_norm_rope(const float* __restrict__ in,
                                                    const float* __restrict__ w,
                                                    const int* __restrict__ pos_ids,
                                                    unsigned short* __restrict__ out, int nheads) {
  size_t row = blockIdx.x;
  int t = threadIdx.x;
  const float* x = in + row * HD_;
  float v = x[t];
  float tot = block_sum256(v * v);
  float inv = rsqrtf(tot / (float)HD_ + 1e-6f);
  int pos = pos_ids[row / nheads];
  int d2 = t & 127;
  float freq = __expf(-((float)(2 * d2) / 256.f) * logf(10000.f));
  float ang = (float)pos * freq;
  float c = __cosf(ang), s = __sinf(ang);
  int partner = (t < 128) ? (t + 128) : (t - 128);
  float xn = (1.f + w[t]) * v * inv;
  float xp = (1.f + w[partner]) * x[partner] * inv;
  float o = (t < 128) ? (xn * c - xp * s) : (xn * c + xp * s);
  out[row * HD_ + t] = f2bf(o);
}

// ---------------- generic WMMA GEMM: C[M,N] = A(bf16)[M,K] x B(fp32->bf16)[K,N] ----------------
// EPI 0: fp32 out; EPI 1: bf16 out; EPI 2: bf16 out = gelu(aux) * acc
template<int EPI>
__global__ __launch_bounds__(256) void gemm_bf16(const unsigned short* __restrict__ A,
                                                 const float* __restrict__ Bw,
                                                 float* __restrict__ Cf,
                                                 unsigned short* __restrict__ Cb,
                                                 const float* __restrict__ aux,
                                                 int M, int N, int K) {
  __shared__ __align__(16) unsigned short sA[128][40];
  __shared__ __align__(16) unsigned short sB[128][40]; // transposed: [n][k]
  int tid = threadIdx.x;
  int wv = tid >> 5, lane = tid & 31;
  int wm = wv & 3, wn = wv >> 2;
  int m0 = blockIdx.y * 128, n0 = blockIdx.x * 128;
  int mi = lane & 15, hi = lane >> 4, koff = hi * 8;

  v8f acc[2][4];
  #pragma unroll
  for (int i = 0; i < 2; ++i)
    #pragma unroll
    for (int j = 0; j < 4; ++j) acc[i][j] = vzero8();

  for (int k0 = 0; k0 < K; k0 += 32) {
    { // stage A tile 128x32 (bf16): async global->LDS, 32B per lane
      int row = tid >> 1, col = (tid & 1) * 16;
      const unsigned short* gp = A + (size_t)(m0 + row) * K + k0 + col;
      async_copy32(lds_off(&sA[row][col]), gp);
    }
    { // stage B tile 32x128 fp32 -> bf16 transposed
      int kk = tid & 31, nb = (tid >> 5) * 16;
      const float* g = Bw + (size_t)(k0 + kk) * N + n0 + nb;
      float4 f0 = ((const float4*)g)[0];
      float4 f1 = ((const float4*)g)[1];
      float4 f2 = ((const float4*)g)[2];
      float4 f3 = ((const float4*)g)[3];
      float fv[16] = {f0.x,f0.y,f0.z,f0.w, f1.x,f1.y,f1.z,f1.w,
                      f2.x,f2.y,f2.z,f2.w, f3.x,f3.y,f3.z,f3.w};
      #pragma unroll
      for (int j = 0; j < 16; ++j) sB[nb + j][kk] = f2bf(fv[j]);
      if (k0 + 32 < K) __builtin_prefetch(Bw + (size_t)(k0 + 32 + kk) * N + n0 + nb, 0, 1);
    }
    wait_async();
    __syncthreads();

    Frag a[2], b[4];
    #pragma unroll
    for (int i = 0; i < 2; ++i) {
      int r = wm * 32 + i * 16 + mi;
      a[i].q[0] = *(const uint4*)&sA[r][koff];
      a[i].q[1] = *(const uint4*)&sA[r][16 + koff];
    }
    #pragma unroll
    for (int j = 0; j < 4; ++j) {
      int c = wn * 64 + j * 16 + mi;
      b[j].q[0] = *(const uint4*)&sB[c][koff];
      b[j].q[1] = *(const uint4*)&sB[c][16 + koff];
    }
    #pragma unroll
    for (int i = 0; i < 2; ++i)
      #pragma unroll
      for (int j = 0; j < 4; ++j)
        acc[i][j] = wmma_bf16(a[i].v, b[j].v, acc[i][j]);
    __syncthreads();
  }

  #pragma unroll
  for (int i = 0; i < 2; ++i)
    #pragma unroll
    for (int j = 0; j < 4; ++j)
      #pragma unroll
      for (int r = 0; r < 8; ++r) {
        int grow = m0 + wm * 32 + i * 16 + r + 8 * hi;
        int gcol = n0 + wn * 64 + j * 16 + mi;
        size_t idx = (size_t)grow * N + gcol;
        float vv = acc[i][j][r];
        if (EPI == 0) Cf[idx] = vv;
        else if (EPI == 1) Cb[idx] = f2bf(vv);
        else { float g = aux[idx]; Cb[idx] = f2bf(gelu_tanh(g) * vv); }
      }
}

// ---------------- flash attention, sliding-window causal, WMMA ----------------
__global__ __launch_bounds__(256) void attn_kernel(const unsigned short* __restrict__ Q,
                                                   const unsigned short* __restrict__ Kb,
                                                   const unsigned short* __restrict__ Vb,
                                                   unsigned short* __restrict__ O) {
  __shared__ __align__(16) unsigned short sK[32][264];   // 32 keys x 256 hd
  __shared__ __align__(16) unsigned short sVT[256][40];  // hd x 32 keys
  __shared__ __align__(16) unsigned short sP[8][16][40]; // per-wave P 16x32
  int tid = threadIdx.x, wv = tid >> 5, lane = tid & 31;
  int head = blockIdx.y, b = blockIdx.z;
  int kvh = head >> 1; // NH/NKV = 2
  int q0blk = blockIdx.x * 128;
  long bS0 = (long)b * SS;
  int qrow0 = q0blk + wv * 16;
  int mi = lane & 15, hi = lane >> 4, koff = hi * 8;

  // Q fragments for this wave's 16 rows (held across the key loop)
  Frag aq[8];
  {
    const unsigned short* qp = Q + (((size_t)(bS0 + qrow0 + mi)) * NH_ + head) * HD_;
    #pragma unroll
    for (int t = 0; t < 8; ++t) {
      aq[t].q[0] = *(const uint4*)(qp + t * 32 + koff);
      aq[t].q[1] = *(const uint4*)(qp + t * 32 + 16 + koff);
    }
  }

  v8f acc[16];
  #pragma unroll
  for (int d = 0; d < 16; ++d) acc[d] = vzero8();
  float rmax[8], rsum[8];
  #pragma unroll
  for (int j = 0; j < 8; ++j) { rmax[j] = -1e30f; rsum[j] = 0.f; }

  int kstart = q0blk - (SW_ - 1); if (kstart < 0) kstart = 0; kstart &= ~31;
  int kend = q0blk + 127;

  for (int c0 = kstart; c0 <= kend; c0 += 32) {
    { // stage K chunk via async global->LDS (64B per lane); V chunk transposed via VALU
      int kk = tid >> 3, cb = (tid & 7) * 32;
      size_t krow = ((size_t)(bS0 + c0 + kk)) * NKV_ + kvh;
      async_copy64(lds_off(&sK[kk][cb]), Kb + krow * HD_ + cb);
      const uint4* gv = (const uint4*)(Vb + krow * HD_ + cb);
      union { uint4 u[4]; unsigned short h[32]; } vt;
      vt.u[0] = gv[0]; vt.u[1] = gv[1]; vt.u[2] = gv[2]; vt.u[3] = gv[3];
      #pragma unroll
      for (int i = 0; i < 32; ++i) sVT[cb + i][kk] = vt.h[i];
    }
    wait_async();
    __syncthreads();

    bool active = (c0 <= qrow0 + 15) && (c0 + 31 >= qrow0 - (SW_ - 1));
    if (active) {
      v8f s0 = vzero8(), s1 = vzero8();
      #pragma unroll
      for (int t = 0; t < 8; ++t) {
        Frag b0, b1;
        b0.q[0] = *(const uint4*)&sK[mi][t * 32 + koff];
        b0.q[1] = *(const uint4*)&sK[mi][t * 32 + 16 + koff];
        b1.q[0] = *(const uint4*)&sK[16 + mi][t * 32 + koff];
        b1.q[1] = *(const uint4*)&sK[16 + mi][t * 32 + 16 + koff];
        s0 = wmma_bf16(aq[t].v, b0.v, s0);
        s1 = wmma_bf16(aq[t].v, b1.v, s1);
      }
      int kt0 = c0 + mi, kt1 = kt0 + 16;
      float scl[8];
      #pragma unroll
      for (int j = 0; j < 8; ++j) {
        int qt = qrow0 + j + 8 * hi;
        float a0 = s0[j] * SCALE_; int r0 = qt - kt0; if (r0 < 0 || r0 >= SW_) a0 = -1e30f;
        float a1 = s1[j] * SCALE_; int r1 = qt - kt1; if (r1 < 0 || r1 >= SW_) a1 = -1e30f;
        float m = fmaxf(a0, a1);
        m = fmaxf(m, __shfl_xor(m, 1));
        m = fmaxf(m, __shfl_xor(m, 2));
        m = fmaxf(m, __shfl_xor(m, 4));
        m = fmaxf(m, __shfl_xor(m, 8));
        float nm = fmaxf(rmax[j], m);
        float sc = __expf(rmax[j] - nm);
        float p0 = __expf(a0 - nm), p1 = __expf(a1 - nm);
        float ps = p0 + p1;
        ps += __shfl_xor(ps, 1);
        ps += __shfl_xor(ps, 2);
        ps += __shfl_xor(ps, 4);
        ps += __shfl_xor(ps, 8);
        rsum[j] = rsum[j] * sc + ps;
        rmax[j] = nm; scl[j] = sc;
        sP[wv][j + 8 * hi][mi]      = f2bf(p0);
        sP[wv][j + 8 * hi][16 + mi] = f2bf(p1);
      }
      #pragma unroll
      for (int d = 0; d < 16; ++d)
        #pragma unroll
        for (int j = 0; j < 8; ++j) acc[d][j] *= scl[j];
      Frag ap;
      ap.q[0] = *(const uint4*)&sP[wv][mi][koff];
      ap.q[1] = *(const uint4*)&sP[wv][mi][16 + koff];
      #pragma unroll
      for (int d = 0; d < 16; ++d) {
        Frag bv; int col = d * 16 + mi;
        bv.q[0] = *(const uint4*)&sVT[col][koff];
        bv.q[1] = *(const uint4*)&sVT[col][16 + koff];
        acc[d] = wmma_bf16(ap.v, bv.v, acc[d]);
      }
    }
    __syncthreads();
  }

  #pragma unroll
  for (int d = 0; d < 16; ++d)
    #pragma unroll
    for (int j = 0; j < 8; ++j) {
      int qt = qrow0 + j + 8 * hi;
      size_t idx = ((size_t)(bS0 + qt)) * (NH_ * HD_) + (size_t)head * HD_ + d * 16 + mi;
      O[idx] = f2bf(acc[d][j] / rsum[j]);
    }
}

extern "C" void kernel_launch(void* const* d_in, const int* in_sizes, int n_in,
                              void* d_out, int out_size, void* d_ws, size_t ws_size,
                              hipStream_t stream) {
  (void)in_sizes; (void)n_in; (void)out_size; (void)ws_size;
  const float* hs      = (const float*)d_in[0];
  const float* wq      = (const float*)d_in[1];
  const float* wk      = (const float*)d_in[2];
  const float* wvw     = (const float*)d_in[3];
  const float* wo      = (const float*)d_in[4];
  const float* w_gate  = (const float*)d_in[5];
  const float* w_up    = (const float*)d_in[6];
  const float* w_down  = (const float*)d_in[7];
  const float* ln_in   = (const float*)d_in[8];
  const float* ln_pa   = (const float*)d_in[9];
  const float* ln_pf   = (const float*)d_in[10];
  const float* ln_pff  = (const float*)d_in[11];
  const float* qn      = (const float*)d_in[12];
  const float* kn      = (const float*)d_in[13];
  const int*   pos     = (const int*)d_in[14];

  char* ws = (char*)d_ws;
  const size_t R = (size_t)BB * SS; // 4096 rows
  auto al = [](size_t x) { return (x + 255) & ~(size_t)255; };
  size_t off = 0;
  size_t xnorm_o = off; off += al(R * HH * 2);
  size_t qf_o    = off; off += al(R * NH_ * HD_ * 4);
  size_t kf_o    = off; off += al(R * NKV_ * HD_ * 4);
  size_t vb_o    = off; off += al(R * NKV_ * HD_ * 2);
  size_t qb_o    = off; off += al(R * NH_ * HD_ * 2);
  size_t kb_o    = off; off += al(R * NKV_ * HD_ * 2);
  size_t at_o    = off; off += al(R * NH_ * HD_ * 2);
  size_t ap_o    = off; off += al(R * HH * 4);
  size_t phase1_end = off;
  size_t gate_o = 0;                       // reuses phase-1 region
  size_t gate_bytes = al(R * (size_t)INTER_ * 4);
  size_t base2 = gate_bytes > phase1_end ? gate_bytes : phase1_end;
  size_t hid_o   = base2;            base2 += al(R * HH * 4);
  size_t x2_o    = base2;            base2 += al(R * HH * 2);
  size_t gated_o = base2;            base2 += al(R * (size_t)INTER_ * 2);
  size_t mlp_o   = base2;

  unsigned short* xnorm = (unsigned short*)(ws + xnorm_o);
  float*          qf    = (float*)(ws + qf_o);
  float*          kf    = (float*)(ws + kf_o);
  unsigned short* vb    = (unsigned short*)(ws + vb_o);
  unsigned short* qb    = (unsigned short*)(ws + qb_o);
  unsigned short* kb    = (unsigned short*)(ws + kb_o);
  unsigned short* atb   = (unsigned short*)(ws + at_o);
  float*          aproj = (float*)(ws + ap_o);
  float*          gate  = (float*)(ws + gate_o);
  float*          hid   = (float*)(ws + hid_o);
  unsigned short* x2    = (unsigned short*)(ws + x2_o);
  unsigned short* gated = (unsigned short*)(ws + gated_o);
  float*          mlp   = (float*)(ws + mlp_o);
  float*          outp  = (float*)d_out;

  // 1) input RMSNorm -> bf16
  rms_to_bf16<<<R, 256, 0, stream>>>(hs, ln_in, xnorm, HH);
  // 2) QKV projections
  gemm_bf16<0><<<dim3((NH_*HD_)/128, R/128), 256, 0, stream>>>(xnorm, wq,  qf, nullptr, nullptr, (int)R, NH_*HD_,  HH);
  gemm_bf16<0><<<dim3((NKV_*HD_)/128, R/128), 256, 0, stream>>>(xnorm, wk,  kf, nullptr, nullptr, (int)R, NKV_*HD_, HH);
  gemm_bf16<1><<<dim3((NKV_*HD_)/128, R/128), 256, 0, stream>>>(xnorm, wvw, nullptr, vb, nullptr, (int)R, NKV_*HD_, HH);
  // 3) QK head RMSNorm + RoPE -> bf16
  qk_norm_rope<<<R * NH_, 256, 0, stream>>>(qf, qn, pos, qb, NH_);
  qk_norm_rope<<<R * NKV_, 256, 0, stream>>>(kf, kn, pos, kb, NKV_);
  // 4) sliding-window flash attention
  attn_kernel<<<dim3(SS/128, NH_, BB), 256, 0, stream>>>(qb, kb, vb, atb);
  // 5) output projection
  gemm_bf16<0><<<dim3(HH/128, R/128), 256, 0, stream>>>(atb, wo, aproj, nullptr, nullptr, (int)R, HH, NH_*HD_);
  // 6) hidden = residual + RMS(attn)
  rms_add<<<R, 256, 0, stream>>>(aproj, ln_pa, hs, hid, HH);
  // 7) pre-FF RMSNorm -> bf16
  rms_to_bf16<<<R, 256, 0, stream>>>(hid, ln_pf, x2, HH);
  // 8) gate GEMM (fp32), then up GEMM with fused gelu(gate)*up -> bf16
  gemm_bf16<0><<<dim3(INTER_/128, R/128), 256, 0, stream>>>(x2, w_gate, gate, nullptr, nullptr, (int)R, INTER_, HH);
  gemm_bf16<2><<<dim3(INTER_/128, R/128), 256, 0, stream>>>(x2, w_up, nullptr, gated, gate, (int)R, INTER_, HH);
  // 9) down GEMM
  gemm_bf16<0><<<dim3(HH/128, R/128), 256, 0, stream>>>(gated, w_down, mlp, nullptr, nullptr, (int)R, HH, INTER_);
  // 10) final residual + RMS
  rms_add<<<R, 256, 0, stream>>>(mlp, ln_pff, hid, outp, HH);
}
